// RefTransformer_44796508897938
// MI455X (gfx1250) — compile-verified, matrix operand order checked by reading, not run on hardware
//
#include <hip/hip_runtime.h>
#include <math.h>

typedef __attribute__((ext_vector_type(16))) __bf16 v16bf;
typedef __attribute__((ext_vector_type(8)))  __bf16 v8bf;
typedef __attribute__((ext_vector_type(8)))  float  v8f;
typedef __attribute__((ext_vector_type(4)))  unsigned int u32x4;
typedef __attribute__((ext_vector_type(8)))  int i32x8;
typedef __attribute__((ext_vector_type(4)))  int i32x4;

#if defined(__HIP_DEVICE_COMPILE__) && defined(__has_builtin)
#  if __has_builtin(__builtin_amdgcn_tensor_load_to_lds) && \
      __has_builtin(__builtin_amdgcn_s_wait_tensorcnt)
#    define USE_TDM 1
#  endif
#endif
#ifndef USE_TDM
#  define USE_TDM 0
#endif

namespace {
constexpr int kB     = 4;
constexpr int kNF    = 16;
constexpr int kH     = 256;
constexpr int kW     = 256;
constexpr int kC     = 1024;   // token dim
constexpr int kInner = 3072;   // 3*C
constexpr int kNpat  = 33 * 33;  // 1089 real tokens
constexpr int kNP    = 1152;     // padded tokens: 18*64 = 9*128 = 36*32
constexpr float kScale = 0.03125f;  // 1024^-0.5
constexpr float kEps   = 1e-5f;

constexpr int EP_F32_ALPHA     = 0;
constexpr int EP_BF16_BIAS     = 1;
constexpr int EP_BF16_GELU     = 2;
constexpr int EP_F32_BIAS_RES  = 3;

constexpr int BM = 64, BN = 128, BK = 32, LDSS = 40;  // LDS row stride (bf16 elems)
}

#if USE_TDM
// ---------------------------------------------------------------------------
// Issue one TDM 2-D tile load: rows x 32 bf16 elems, row stride = ld elems,
// into LDS at lds_byte with hardware row padding 64B data + 16B pad (LDSS=40).
// D# group0/group1 packed per CDNA5 ISA.
// Toolchain uses the 6-arg builtin: (g0, g1, g2, g3, g4, cpol).
// ---------------------------------------------------------------------------
__device__ __forceinline__ void tdm_load_tile(const __bf16* gsrc,
                                              unsigned lds_byte,
                                              int rows, long ld,
                                              long rem_rows, long rem_cols) {
  const unsigned long long ga = (unsigned long long)(uintptr_t)gsrc;
  u32x4 g0;
  g0[0] = 1u;                                   // count=1, user mode
  g0[1] = lds_byte;                             // lds_addr
  g0[2] = (unsigned)ga;                         // global_addr[31:0]
  g0[3] = (unsigned)(ga >> 32) | (2u << 30);    // global_addr[56:32] | type=2

  const unsigned td0 = (unsigned)rem_cols;      // tensor_dim0 (OOB bound)
  const unsigned td1 = (unsigned)rem_rows;      // tensor_dim1 (OOB bound)
  const unsigned long long st = (unsigned long long)ld;  // dim0 stride (elems)
  i32x8 g1;
  // data_size=1 (2B) | pad_enable | pad_interval=3 (16 DW) | pad_amount=3 (4 DW)
  g1[0] = (int)((1u << 16) | (1u << 20) | (3u << 22) | (3u << 25));
  g1[1] = (int)((td0 & 0xFFFFu) << 16);                       // td0[15:0]
  g1[2] = (int)((td0 >> 16) | ((td1 & 0xFFFFu) << 16));       // td0[31:16], td1[15:0]
  g1[3] = (int)((td1 >> 16) | ((unsigned)BK << 16));          // td1[31:16], tile_dim0=32
  g1[4] = rows;                                               // tile_dim1; tile_dim2=0
  g1[5] = (int)(unsigned)st;                                  // dim0_stride[31:0]
  g1[6] = (int)(unsigned)(st >> 32);                          // dim0_stride[47:32]
  g1[7] = 0;

  const i32x4 z4 = {0, 0, 0, 0};
  const i32x8 z8 = {0, 0, 0, 0, 0, 0, 0, 0};
  __builtin_amdgcn_tensor_load_to_lds(g0, g1, z4, z4, z8, 0);
}
#endif

// ---------------------------------------------------------------------------
// Generic GEMM:  C[M,N] = A[M,K] * Bt[N,K]^T   (Bt stored row-major [N,K])
// bf16 inputs, f32 WMMA accumulation, TDM-fed double-buffered LDS staging.
// ---------------------------------------------------------------------------
__global__ __launch_bounds__(256)
void gemm_bf16_wmma(const __bf16* __restrict__ A, long sAb,
                    const __bf16* __restrict__ Bt, long sBb,
                    void* __restrict__ Cptr, long sCb,
                    const float* __restrict__ bias,
                    const float* __restrict__ res, long sRb,
                    int M, int N, int K, float alpha, int ep) {
  __shared__ __bf16 sA[2][BM * LDSS];
  __shared__ __bf16 sB[2][BN * LDSS];

  const int bz   = blockIdx.z;
  const __bf16* Ab  = A  + (long)bz * sAb;
  const __bf16* Btb = Bt + (long)bz * sBb;
  float*  Cf = (float*)Cptr  + (long)bz * sCb;
  __bf16* Ch = (__bf16*)Cptr + (long)bz * sCb;
  const float* resb = res ? res + (long)bz * sRb : nullptr;

  const int tid  = threadIdx.x;
  const int wave = tid >> 5;
  const int lane = tid & 31;
  const int m0 = blockIdx.y * BM;
  const int n0 = blockIdx.x * BN;
  const int wm = (wave >> 2) * 32;   // 0 / 32
  const int wn = (wave & 3) * 32;    // 0..96

  const __bf16* Atile = Ab  + (long)m0 * K;
  const __bf16* Btile = Btb + (long)n0 * K;

  v8f acc[2][2];
#pragma unroll
  for (int s = 0; s < 2; ++s)
#pragma unroll
    for (int t = 0; t < 2; ++t) acc[s][t] = (v8f)(0.f);

  const int lrow = lane & 15;          // M (or N) index within 16
  const int kgrp = (lane >> 4) * 8;    // K sub-group base: 0 or 8
  const int nk = K / BK;

#if USE_TDM
  if (wave == 0) {  // prologue: DMA tile 0 into buffer 0
    tdm_load_tile(Atile, (unsigned)(uintptr_t)&sA[0][0], BM, K, M - m0, K);
    tdm_load_tile(Btile, (unsigned)(uintptr_t)&sB[0][0], BN, K, N - n0, K);
  }
#endif

  for (int kt = 0; kt < nk; ++kt) {
    const int buf = kt & 1;
    const int k0 = kt * BK;
#if USE_TDM
    if (wave == 0) {
      if (kt + 1 < nk) {
        // prefetch next tile pair into the other buffer (readers of that
        // buffer finished at the end-of-iteration barrier of kt-1)
        tdm_load_tile(Atile + (k0 + BK), (unsigned)(uintptr_t)&sA[buf ^ 1][0],
                      BM, K, M - m0, K - (k0 + BK));
        tdm_load_tile(Btile + (k0 + BK), (unsigned)(uintptr_t)&sB[buf ^ 1][0],
                      BN, K, N - n0, K - (k0 + BK));
        // TDM completes in order per wave: <=2 outstanding => tile kt landed
        __builtin_amdgcn_s_wait_tensorcnt(2);
      } else {
        __builtin_amdgcn_s_wait_tensorcnt(0);
      }
    }
#else
    {
      const int ldr = tid >> 2;          // 0..63
      const int ldg = (tid & 3) * 8;     // 0,8,16,24
      *(uint4*)&sA[buf][ldr * LDSS + ldg] =
          *(const uint4*)(Atile + (long)ldr * K + k0 + ldg);
      *(uint4*)&sB[buf][ldr * LDSS + ldg] =
          *(const uint4*)(Btile + (long)ldr * K + k0 + ldg);
      *(uint4*)&sB[buf][(ldr + 64) * LDSS + ldg] =
          *(const uint4*)(Btile + (long)(ldr + 64) * K + k0 + ldg);
    }
#endif
    __syncthreads();   // publish tile kt to all waves

    v16bf af[2], bfr[2];
#pragma unroll
    for (int s = 0; s < 2; ++s) {
      const __bf16* p = &sA[buf][(wm + s * 16 + lrow) * LDSS + kgrp];
      v8bf lo = *(const v8bf*)p;
      v8bf hi = *(const v8bf*)(p + 16);
      af[s] = __builtin_shufflevector(lo, hi, 0,1,2,3,4,5,6,7,8,9,10,11,12,13,14,15);
    }
#pragma unroll
    for (int t = 0; t < 2; ++t) {
      const __bf16* p = &sB[buf][(wn + t * 16 + lrow) * LDSS + kgrp];
      v8bf lo = *(const v8bf*)p;
      v8bf hi = *(const v8bf*)(p + 16);
      bfr[t] = __builtin_shufflevector(lo, hi, 0,1,2,3,4,5,6,7,8,9,10,11,12,13,14,15);
    }
#pragma unroll
    for (int s = 0; s < 2; ++s)
#pragma unroll
      for (int t = 0; t < 2; ++t)
        acc[s][t] = __builtin_amdgcn_wmma_f32_16x16x32_bf16(
            false, af[s], false, bfr[t], (short)0, acc[s][t], false, false);
    __syncthreads();   // all reads of buf done before it is refilled
  }

  // epilogue: D layout — lane 0..15: N=lane, M=v ; lane 16..31: N=lane-16, M=v+8
  const int cn = lane & 15;
  const int mo = (lane >> 4) * 8;
#pragma unroll
  for (int s = 0; s < 2; ++s)
#pragma unroll
    for (int t = 0; t < 2; ++t) {
      const int gn = n0 + wn + t * 16 + cn;
#pragma unroll
      for (int v = 0; v < 8; ++v) {
        const int gm = m0 + wm + s * 16 + mo + v;
        const long off = (long)gm * N + gn;
        const float val = acc[s][t][v];
        if (ep == EP_F32_ALPHA) {
          Cf[off] = val * alpha;
        } else if (ep == EP_BF16_BIAS) {
          const float bb = bias ? bias[gn] : 0.f;
          Ch[off] = (__bf16)(val + bb);
        } else if (ep == EP_BF16_GELU) {
          const float u = val + bias[gn];
          Ch[off] = (__bf16)(0.5f * u * (1.f + erff(u * 0.70710678118654752f)));
        } else {  // EP_F32_BIAS_RES
          Cf[off] = val + bias[gn] + resb[off];
        }
      }
    }
}

// ---------------------------------------------------------------------------
// Patchify with reflect pad: img[B,16,256,256] -> tok[B,NP,1024] (f32, zero pad)
// feature d = c*64 + kh*8 + kw ; n = py*33 + px ; p = shift*2
// ---------------------------------------------------------------------------
__global__ __launch_bounds__(256)
void patch_kernel(const float* __restrict__ img, float* __restrict__ tok, int shift) {
  const long idx = (long)blockIdx.x * 256 + threadIdx.x;   // B*NP*C
  const int d = (int)(idx & (kC - 1));
  const long t = idx >> 10;
  const int n = (int)(t % kNP);
  const int b = (int)(t / kNP);
  if (n >= kNpat) { tok[idx] = 0.f; return; }
  const int py = n / 33, px = n % 33;
  const int c = d >> 6, kh = (d >> 3) & 7, kw = d & 7;
  const int p = shift * 2;
  int h = py * 8 + kh - p;
  int w = px * 8 + kw - p;
  if (h < 0) h = -h;
  if (h >= kH) h = 2 * kH - 2 - h;
  if (w < 0) w = -w;
  if (w >= kW) w = 2 * kW - 2 - w;
  tok[idx] = img[(((long)b * kNF + c) * kH + h) * kW + w];
}

// ---------------------------------------------------------------------------
// Unpatchify: x[B,NP,1024] -> out[B,16,256,256]
// ---------------------------------------------------------------------------
__global__ __launch_bounds__(256)
void unpatch_kernel(const float* __restrict__ x, float* __restrict__ out) {
  const long idx = (long)blockIdx.x * 256 + threadIdx.x;   // B*16*256*256
  const int w = (int)(idx & 255);
  const int h = (int)((idx >> 8) & 255);
  const int c = (int)((idx >> 16) & 15);
  const int b = (int)(idx >> 20);
  const int n = (h >> 3) * 33 + (w >> 3);
  const int d = c * 64 + (h & 7) * 8 + (w & 7);
  out[idx] = x[((long)b * kNP + n) * kC + d];
}

// ---------------------------------------------------------------------------
// LayerNorm over last dim (1024): f32 in -> bf16 out
// ---------------------------------------------------------------------------
__global__ __launch_bounds__(256)
void ln_kernel(const float* __restrict__ x, __bf16* __restrict__ y,
               const float* __restrict__ g, const float* __restrict__ bt) {
  __shared__ float red[256];
  const long row = blockIdx.x;
  const float* xr = x + row * (long)kC;
  const int tid = threadIdx.x;
  float v[4];
  float s = 0.f;
#pragma unroll
  for (int i = 0; i < 4; ++i) { v[i] = xr[tid + 256 * i]; s += v[i]; }
  red[tid] = s; __syncthreads();
  for (int o = 128; o > 0; o >>= 1) { if (tid < o) red[tid] += red[tid + o]; __syncthreads(); }
  const float mean = red[0] * (1.f / kC);
  __syncthreads();
  float s2 = 0.f;
#pragma unroll
  for (int i = 0; i < 4; ++i) { const float dd = v[i] - mean; s2 += dd * dd; }
  red[tid] = s2; __syncthreads();
  for (int o = 128; o > 0; o >>= 1) { if (tid < o) red[tid] += red[tid + o]; __syncthreads(); }
  const float rstd = rsqrtf(red[0] * (1.f / kC) + kEps);
#pragma unroll
  for (int i = 0; i < 4; ++i) {
    const int c = tid + 256 * i;
    y[row * (long)kC + c] = (__bf16)((v[i] - mean) * rstd * g[c] + bt[c]);
  }
}

// ---------------------------------------------------------------------------
// Masked row softmax: S[NP] f32 -> P[NP] bf16 ; cols >= kNpat masked to 0
// ---------------------------------------------------------------------------
__global__ __launch_bounds__(256)
void softmax_kernel(const float* __restrict__ S, __bf16* __restrict__ P) {
  __shared__ float red[256];
  const long row = blockIdx.x;
  const float* Sr = S + row * (long)kNP;
  __bf16* Pr = P + row * (long)kNP;
  const int tid = threadIdx.x;
  float mx = -3.4e38f;
  for (int c = tid; c < kNpat; c += 256) mx = fmaxf(mx, Sr[c]);
  red[tid] = mx; __syncthreads();
  for (int o = 128; o > 0; o >>= 1) { if (tid < o) red[tid] = fmaxf(red[tid], red[tid + o]); __syncthreads(); }
  mx = red[0]; __syncthreads();
  float sum = 0.f;
  for (int c = tid; c < kNpat; c += 256) sum += expf(Sr[c] - mx);
  red[tid] = sum; __syncthreads();
  for (int o = 128; o > 0; o >>= 1) { if (tid < o) red[tid] += red[tid + o]; __syncthreads(); }
  const float rinv = 1.f / red[0];
  for (int c = tid; c < kNP; c += 256)
    Pr[c] = (c < kNpat) ? (__bf16)(expf(Sr[c] - mx) * rinv) : (__bf16)(0.f);
}

// ---------------------------------------------------------------------------
// bf16 batched transpose: in [R,Cc] -> out [Cc,R]
// ---------------------------------------------------------------------------
__global__ __launch_bounds__(256)
void transpose_bf16(const __bf16* __restrict__ in, __bf16* __restrict__ out,
                    int R, int Cc, long sIn, long sOut) {
  __shared__ __bf16 t[32][33];
  const int b = blockIdx.z;
  const int r0 = blockIdx.y * 32, c0 = blockIdx.x * 32;
  const int tx = threadIdx.x & 31, ty = threadIdx.x >> 5;
  const __bf16* ib = in + (long)b * sIn;
  __bf16* ob = out + (long)b * sOut;
#pragma unroll
  for (int i = 0; i < 32; i += 8)
    t[ty + i][tx] = ib[(long)(r0 + ty + i) * Cc + c0 + tx];
  __syncthreads();
#pragma unroll
  for (int i = 0; i < 32; i += 8)
    ob[(long)(c0 + ty + i) * R + r0 + tx] = t[tx][ty + i];
}

// ---------------------------------------------------------------------------
// Weight convert+transpose: W[K,N] f32 -> Wt[N,K] bf16
// ---------------------------------------------------------------------------
__global__ __launch_bounds__(256)
void wt_convert(const float* __restrict__ W, __bf16* __restrict__ Wt, int K, int N) {
  const long idx = (long)blockIdx.x * 256 + threadIdx.x;  // N*K
  const int k = (int)(idx % K);
  const int n = (int)(idx / K);
  Wt[idx] = (__bf16)W[(long)k * N + n];
}

// ---------------------------------------------------------------------------
extern "C" void kernel_launch(void* const* d_in, const int* in_sizes, int n_in,
                              void* d_out, int out_size, void* d_ws, size_t ws_size,
                              hipStream_t stream) {
  (void)in_sizes; (void)n_in; (void)out_size; (void)ws_size;
  const float* dr_img = (const float*)d_in[0];
  const float* dr_ref = (const float*)d_in[1];
  const float* cl_ref = (const float*)d_in[2];
  const float* ln_g = (const float*)d_in[3];
  const float* ln_b = (const float*)d_in[4];
  const float* Wq = (const float*)d_in[5];  const float* bq = (const float*)d_in[6];
  const float* Wk = (const float*)d_in[7];  const float* bk = (const float*)d_in[8];
  const float* Wv = (const float*)d_in[9];  const float* bv = (const float*)d_in[10];
  const float* Wo = (const float*)d_in[11]; const float* bo = (const float*)d_in[12];
  const float* ff_g = (const float*)d_in[13];
  const float* ff_b = (const float*)d_in[14];
  const float* W1 = (const float*)d_in[15]; const float* b1 = (const float*)d_in[16];
  const float* W2 = (const float*)d_in[17]; const float* b2 = (const float*)d_in[18];
  float* out = (float*)d_out;

  char* p = (char*)d_ws;
  auto alloc = [&](size_t bytes) -> char* {
    char* r = p; p += (bytes + 255) & ~(size_t)255; return r;
  };
  const size_t tokF = (size_t)kB * kNP * kC * 4;        // f32 token buf
  const size_t tokH = (size_t)kB * kNP * kC * 2;        // bf16 token buf
  const size_t innH = (size_t)kB * kNP * kInner * 2;    // bf16 inner buf
  float*  x0     = (float*)alloc(tokF);
  float*  x1     = (float*)alloc(tokF);
  float*  ktok   = (float*)alloc(tokF);
  float*  vtok   = (float*)alloc(tokF);
  __bf16* lnbuf  = (__bf16*)alloc(tokH);
  __bf16* gbuf   = (__bf16*)alloc(tokH);
  __bf16* qp     = (__bf16*)alloc(innH);
  __bf16* kp     = (__bf16*)alloc(innH);
  __bf16* vp     = (__bf16*)alloc(innH);
  __bf16* vpT    = (__bf16*)alloc(innH);
  float*  scores = (float*)alloc((size_t)kB * kNP * kNP * 4);
  __bf16* probs  = (__bf16*)alloc((size_t)kB * kNP * kNP * 2);
  __bf16* av     = (__bf16*)alloc(innH);
  __bf16* WqT    = (__bf16*)alloc((size_t)kC * kInner * 2);
  __bf16* WkT    = (__bf16*)alloc((size_t)kC * kInner * 2);
  __bf16* WvT    = (__bf16*)alloc((size_t)kC * kInner * 2);
  __bf16* WoT    = (__bf16*)alloc((size_t)kC * kInner * 2);
  __bf16* W1T    = (__bf16*)alloc((size_t)kC * kC * 2);
  __bf16* W2T    = (__bf16*)alloc((size_t)kC * kC * 2);

  const long sTok = (long)kNP * kC;      // elem strides per batch
  const long sInn = (long)kNP * kInner;
  const long sSco = (long)kNP * kNP;

  // ---- one-time weight conversion (per launch; deterministic) ----
  wt_convert<<<(kC * kInner) / 256, 256, 0, stream>>>(Wq, WqT, kC, kInner);
  wt_convert<<<(kC * kInner) / 256, 256, 0, stream>>>(Wk, WkT, kC, kInner);
  wt_convert<<<(kC * kInner) / 256, 256, 0, stream>>>(Wv, WvT, kC, kInner);
  wt_convert<<<(kC * kInner) / 256, 256, 0, stream>>>(Wo, WoT, kInner, kC);
  wt_convert<<<(kC * kC) / 256, 256, 0, stream>>>(W1, W1T, kC, kC);
  wt_convert<<<(kC * kC) / 256, 256, 0, stream>>>(W2, W2T, kC, kC);

  const int nTokBlk = (kB * kNP * kC) / 256;
  patch_kernel<<<nTokBlk, 256, 0, stream>>>(dr_img, x0, 0);

  const dim3 gQKV(kInner / BN, kNP / BM, kB);   // 24 x 18 x 4
  const dim3 gSco(kNP / BN, kNP / BM, kB);      // 9  x 18 x 4
  const dim3 gOut(kC / BN, kNP / BM, kB);       // 8  x 18 x 4
  const dim3 gTr(kInner / 32, kNP / 32, kB);    // 96 x 36 x 4

  for (int shift = 0; shift < 4; ++shift) {
    patch_kernel<<<nTokBlk, 256, 0, stream>>>(dr_ref, ktok, shift);
    patch_kernel<<<nTokBlk, 256, 0, stream>>>(cl_ref, vtok, shift);

    // qp = LN(x) @ Wq + bq
    ln_kernel<<<kB * kNP, 256, 0, stream>>>(x0, lnbuf, ln_g, ln_b);
    gemm_bf16_wmma<<<gQKV, 256, 0, stream>>>(lnbuf, sTok, WqT, 0, qp, sInn,
        bq, nullptr, 0, kNP, kInner, kC, 1.f, EP_BF16_BIAS);
    // kp = LN(k) @ Wk + bk
    ln_kernel<<<kB * kNP, 256, 0, stream>>>(ktok, lnbuf, ln_g, ln_b);
    gemm_bf16_wmma<<<gQKV, 256, 0, stream>>>(lnbuf, sTok, WkT, 0, kp, sInn,
        bk, nullptr, 0, kNP, kInner, kC, 1.f, EP_BF16_BIAS);
    // vp = LN(v) @ Wv + bv
    ln_kernel<<<kB * kNP, 256, 0, stream>>>(vtok, lnbuf, ln_g, ln_b);
    gemm_bf16_wmma<<<gQKV, 256, 0, stream>>>(lnbuf, sTok, WvT, 0, vp, sInn,
        bv, nullptr, 0, kNP, kInner, kC, 1.f, EP_BF16_BIAS);

    // scores = (qp @ kp^T) * scale   (kp row-major == [N,K] operand)
    gemm_bf16_wmma<<<gSco, 256, 0, stream>>>(qp, sInn, kp, sInn, scores, sSco,
        nullptr, nullptr, 0, kNP, kNP, kInner, kScale, EP_F32_ALPHA);
    softmax_kernel<<<kB * kNP, 256, 0, stream>>>(scores, probs);

    // av = probs @ vp  (needs vp^T as [N,K] operand)
    transpose_bf16<<<gTr, 256, 0, stream>>>(vp, vpT, kNP, kInner, sInn, sInn);
    gemm_bf16_wmma<<<gQKV, 256, 0, stream>>>(probs, sSco, vpT, sInn, av, sInn,
        nullptr, nullptr, 0, kNP, kInner, kNP, 1.f, EP_BF16_BIAS);

    // x1 = av @ Wo + bo + x0
    gemm_bf16_wmma<<<gOut, 256, 0, stream>>>(av, sInn, WoT, 0, x1, sTok,
        bo, x0, sTok, kNP, kC, kInner, 1.f, EP_F32_BIAS_RES);

    // FF: x0 = gelu(LN(x1) @ W1 + b1) @ W2 + b2 + x1
    ln_kernel<<<kB * kNP, 256, 0, stream>>>(x1, lnbuf, ff_g, ff_b);
    gemm_bf16_wmma<<<gOut, 256, 0, stream>>>(lnbuf, sTok, W1T, 0, gbuf, sTok,
        b1, nullptr, 0, kNP, kC, kC, 1.f, EP_BF16_GELU);
    gemm_bf16_wmma<<<gOut, 256, 0, stream>>>(gbuf, sTok, W2T, 0, x0, sTok,
        b2, x1, sTok, kNP, kC, kC, 1.f, EP_F32_BIAS_RES);
  }

  unpatch_kernel<<<(kB * kNF * kH * kW) / 256, 256, 0, stream>>>(x0, out);
}
